// nnue_40587440947549
// MI455X (gfx1250) — matrix operand order repaired and checked by reading, not compiled, not run on hardware
//
#include <hip/hip_runtime.h>

// ---------------------------------------------------------------------------
// NNUE forward for MI455X (gfx1250)
//   k0: W_ft f32 -> bf16 pre-convert (runs each call, ~3us at 23.3 TB/s)
//   k1: bf16 WMMA feature-transformer GEMM, async-to-LDS double-buffered B
//   k2: MLP tail (512->32->32->1) + psqt
// ---------------------------------------------------------------------------

typedef __attribute__((ext_vector_type(16))) __bf16        v16bf;
typedef __attribute__((ext_vector_type(8)))  float         v8f;
typedef __attribute__((ext_vector_type(8)))  unsigned int  v8u;

#define FT_IN    41600                 // 64*64*10 + 64*10
#define BATCH    4096
#define NPAD     272                   // 257 cols padded to 17 * 16
#define WROWS    257                   // KHALF + 1
#define KCHUNKS  2
#define KCHUNK   (FT_IN / KCHUNKS)     // 20800
#define KSTEPS   (KCHUNK / 32)         // 650
#define WPITCH_B (FT_IN * 2)           // bf16 row pitch of Wbf in bytes (83200)
#define BTILE_B  (NPAD * 64)           // one B tile in LDS: 272 rows x 64 B = 17408
#define WBF_BYTES ((size_t)NPAD * FT_IN * 2)   // 22,630,400

__device__ __forceinline__ unsigned int pk_bf16(float a, float b) {
  union { __bf16 h[2]; unsigned int u; } x;
  x.h[0] = (__bf16)a;
  x.h[1] = (__bf16)b;
  return x.u;
}

__device__ __forceinline__ float clamp01(float x) {
  return fminf(fmaxf(x, 0.0f), 1.0f);
}

// CDNA5 async global->LDS copy, 16 B per active lane (ASYNCcnt-tracked).
__device__ __forceinline__ void async_b128(unsigned int lds_off, const char* g) {
  asm volatile("global_load_async_to_lds_b128 %0, %1, off"
               :: "v"(lds_off), "v"(g)
               : "memory");
}

// ---------------------------------------------------------------------------
// Kernel 0: W_ft (257 x 41600 f32) -> Wbf (272 x 41600 bf16, rows 257.. = 0)
// ---------------------------------------------------------------------------
__global__ void __launch_bounds__(256) wconv(const float* __restrict__ W_ft,
                                             unsigned short* __restrict__ Wbf) {
  const int n = blockIdx.x;
  uint2* __restrict__ dst = (uint2*)(Wbf + (size_t)n * FT_IN);
  if (n < WROWS) {
    const float4* __restrict__ src = (const float4*)(W_ft + (size_t)n * FT_IN);
    for (int i = threadIdx.x; i < FT_IN / 4; i += 256) {
      const float4 w = src[i];
      dst[i] = make_uint2(pk_bf16(w.x, w.y), pk_bf16(w.z, w.w));
    }
  } else {
    for (int i = threadIdx.x; i < FT_IN / 4; i += 256)
      dst[i] = make_uint2(0u, 0u);
  }
}

// ---------------------------------------------------------------------------
// Kernel 1: partial GEMM  ftpart[part][row][col] = X_chunk @ W_ft_chunk^T
// Grid: 64 M-tiles * 2 sets * 2 K-chunks = 256 blocks of 256 threads (8 waves)
// B tiles stream into LDS via global_load_async_to_lds_b128 (double-buffered).
// Every thread issues exactly 6 async b128 per step -> s_wait_asynccnt 6 is
// wave-uniform (rows 256..271 are redundantly copied by all waves, same data).
// ---------------------------------------------------------------------------
__device__ __forceinline__ void stage_B(const char* __restrict__ Wb, int k0,
                                        unsigned int lds_base, int t, int lane) {
  // own row n = t (0..255)
  const char* g0 = Wb + (size_t)t * WPITCH_B + (size_t)k0 * 2;
  const unsigned int l0 = lds_base + t * 64;
#pragma unroll
  for (int j = 0; j < 4; ++j) async_b128(l0 + j * 16, g0 + j * 16);
  // pad rows 256..271: 16 rows x 4 chunks = 2 per lane (uniform per wave)
  const int r2 = 256 + (lane & 15);
  const int j2 = (lane >> 4) * 2;
  const char* g1 = Wb + (size_t)r2 * WPITCH_B + (size_t)k0 * 2 + j2 * 16;
  const unsigned int l1 = lds_base + r2 * 64 + j2 * 16;
  async_b128(l1, g1);
  async_b128(l1 + 16, g1 + 16);
}

__global__ void __launch_bounds__(256) ft_gemm(const float* __restrict__ f1,
                                               const float* __restrict__ f2,
                                               const char* __restrict__ Wbf,
                                               float* __restrict__ ftpart) {
  __shared__ uint4 ldsB[2 * NPAD * 4];   // double-buffered: 2 x 17408 B

  const int t    = threadIdx.x;
  const int lane = t & 31;
  const int wave = t >> 5;

  const int b     = blockIdx.x;
  const int mt    = b & 63;
  const int set   = (b >> 6) & 1;
  const int kc    = b >> 7;
  const int Mbase = mt * 64;
  const int part  = kc * 2 + set;
  const int kbase = kc * KCHUNK;

  const float* __restrict__ X = set ? f2 : f1;

  // A-fragment addressing (16-bit A-matrix layout, ISA 7.12.2)
  const int mrow = Mbase + (wave & 3) * 16 + (lane & 15);
  const int koff = (lane >> 4) * 8;
  const float* __restrict__ Xrow = X + (size_t)mrow * FT_IN;

  const int nt0 = (wave < 4) ? 0 : 8;

  const unsigned int lds0 = (unsigned int)(size_t)(void*)&ldsB[0];

  v8f acc[9];
#pragma unroll
  for (int i = 0; i < 9; ++i) acc[i] = (v8f)0.0f;

  // prologue: stage step 0 into buffer 0
  stage_B(Wbf, kbase, lds0, t, lane);

  for (int ks = 0; ks < KSTEPS; ++ks) {
    const int k0  = kbase + ks * 32;
    const int cur = ks & 1;

    // stage next step into the other buffer (overlaps with this step's math)
    if (ks + 1 < KSTEPS) {
      stage_B(Wbf, k0 + 32, lds0 + ((ks + 1) & 1) * BTILE_B, t, lane);
      asm volatile("s_wait_asynccnt 0x6" ::: "memory");  // step-ks copies done
    } else {
      asm volatile("s_wait_asynccnt 0x0" ::: "memory");
    }
    __syncthreads();   // all waves' step-ks tile visible

    // ---- A fragment: direct global load + f32->bf16 convert ---------------
    const float4* pa = (const float4*)(Xrow + k0 + koff);
    const float4* pc = (const float4*)(Xrow + k0 + 16 + koff);
    const float4 a0 = pa[0], a1 = pa[1];
    const float4 c0 = pc[0], c1 = pc[1];
    __builtin_prefetch(Xrow + k0 + 128, 0, 1);  // global_prefetch_b8

    v16bf A;
    A[0]  = (__bf16)a0.x;  A[1]  = (__bf16)a0.y;
    A[2]  = (__bf16)a0.z;  A[3]  = (__bf16)a0.w;
    A[4]  = (__bf16)a1.x;  A[5]  = (__bf16)a1.y;
    A[6]  = (__bf16)a1.z;  A[7]  = (__bf16)a1.w;
    A[8]  = (__bf16)c0.x;  A[9]  = (__bf16)c0.y;
    A[10] = (__bf16)c0.z;  A[11] = (__bf16)c0.w;
    A[12] = (__bf16)c1.x;  A[13] = (__bf16)c1.y;
    A[14] = (__bf16)c1.z;  A[15] = (__bf16)c1.w;

    // ---- load all 9 B fragments first, then 9 back-to-back WMMAs ----------
    const uint4* __restrict__ buf = &ldsB[cur * (NPAD * 4)];
    v16bf Bf[9];
#pragma unroll
    for (int i = 0; i < 9; ++i) {
      const int nIdx = (nt0 + i) * 16 + (lane & 15);
      const uint4* bp = &buf[nIdx * 4 + (lane >> 4)];
      const uint4 u0 = bp[0];
      const uint4 u1 = bp[2];
      const v8u uv = (v8u){u0.x, u0.y, u0.z, u0.w, u1.x, u1.y, u1.z, u1.w};
      Bf[i] = __builtin_bit_cast(v16bf, uv);
    }
#pragma unroll
    for (int i = 0; i < 9; ++i)
      acc[i] = __builtin_amdgcn_wmma_f32_16x16x32_bf16(
          false, A, false, Bf[i], (short)0, acc[i], false, false);

    __syncthreads();   // everyone done reading buf[cur] before it is refilled
  }

  // ---- write partials (disjoint per part; wave w / w+4 tile-8 overlap
  //      stores identical values) --------------------------------------------
  float* __restrict__ O = ftpart + (size_t)part * BATCH * NPAD;
  const int rbase = Mbase + (wave & 3) * 16 + 8 * (lane >> 4);
  const int cbase = lane & 15;
#pragma unroll
  for (int i = 0; i < 9; ++i) {
    const int col = (nt0 + i) * 16 + cbase;
#pragma unroll
    for (int r = 0; r < 8; ++r)
      O[(size_t)(rbase + r) * NPAD + col] = acc[i][r];
  }
}

// ---------------------------------------------------------------------------
// Kernel 2: MLP tail. One wave (32 lanes) per batch row; 8 rows per block.
// ---------------------------------------------------------------------------
__global__ void __launch_bounds__(256) nnue_mlp(const float* __restrict__ P,
                                                const float* __restrict__ b_ft,
                                                const float* __restrict__ W1,
                                                const float* __restrict__ b1,
                                                const float* __restrict__ W2,
                                                const float* __restrict__ b2,
                                                const float* __restrict__ Wo,
                                                const float* __restrict__ bo,
                                                float* __restrict__ out) {
  __shared__ float ldsFt[8 * 512];
  __shared__ float ldsH[8 * 32];

  const int lane = threadIdx.x & 31;
  const int wave = threadIdx.x >> 5;
  const int row  = blockIdx.x * 8 + wave;

  auto ld = [&](int part, int col) {
    return P[((size_t)part * BATCH + row) * NPAD + col];
  };

  float* ftw = ldsFt + wave * 512;
#pragma unroll
  for (int j = 0; j < 8; ++j) {
    const int c = lane + j * 32;
    const float v1 = ld(0, c) + ld(2, c) + b_ft[c];   // set 0, chunks 0+1
    const float v2 = ld(1, c) + ld(3, c) + b_ft[c];   // set 1, chunks 0+1
    ftw[c]       = clamp01(v1);
    ftw[256 + c] = clamp01(v2);
  }
  const float psqt1 = ld(0, 256) + ld(2, 256) + b_ft[256];
  const float psqt2 = ld(1, 256) + ld(3, 256) + b_ft[256];
  __syncthreads();

  // h1[lane] = clamp(W1[lane,:] . ft_out + b1[lane])
  float s = b1[lane];
  {
    const float4* w1r = (const float4*)(W1 + (size_t)lane * 512);
    const float4* fr  = (const float4*)ftw;
#pragma unroll 8
    for (int k = 0; k < 128; ++k) {
      const float4 w = w1r[k];
      const float4 f = fr[k];
      s += w.x * f.x + w.y * f.y + w.z * f.z + w.w * f.w;
    }
  }
  ldsH[wave * 32 + lane] = clamp01(s);
  __syncthreads();

  // h2[lane] = clamp(W2[lane,:] . h1 + b2[lane])
  float s2 = b2[lane];
#pragma unroll
  for (int k = 0; k < 32; ++k)
    s2 += W2[lane * 32 + k] * ldsH[wave * 32 + k];
  s2 = clamp01(s2);

  // out = Wo . h2 + bo + (16 * 1/64) * 0.5 * (psqt1 - psqt2)
  float o = s2 * Wo[lane];
#pragma unroll
  for (int off = 16; off > 0; off >>= 1)
    o += __shfl_xor(o, off, 32);

  if (lane == 0)
    out[row] = o + bo[0] + 0.125f * (psqt1 - psqt2);
}

// ---------------------------------------------------------------------------
extern "C" void kernel_launch(void* const* d_in, const int* in_sizes, int n_in,
                              void* d_out, int out_size, void* d_ws, size_t ws_size,
                              hipStream_t stream) {
  const float* f1   = (const float*)d_in[0];
  const float* f2   = (const float*)d_in[1];
  const float* W_ft = (const float*)d_in[2];
  const float* b_ft = (const float*)d_in[3];
  const float* W1   = (const float*)d_in[4];
  const float* b1   = (const float*)d_in[5];
  const float* W2   = (const float*)d_in[6];
  const float* b2   = (const float*)d_in[7];
  const float* Wo   = (const float*)d_in[8];
  const float* bo   = (const float*)d_in[9];
  float* out = (float*)d_out;

  unsigned short* Wbf = (unsigned short*)d_ws;                 // 22.6 MB bf16
  float* ftpart = (float*)((char*)d_ws + WBF_BYTES);           // 17.8 MB

  wconv  <<<NPAD,      256, 0, stream>>>(W_ft, Wbf);
  ft_gemm<<<256,       256, 0, stream>>>(f1, f2, (const char*)Wbf, ftpart);
  nnue_mlp<<<BATCH / 8, 256, 0, stream>>>(ftpart, b_ft, W1, b1, W2, b2, Wo, bo, out);
}
